// FlashHSAInnerXHierarchicalSparseAttention_6725918785911
// MI455X (gfx1250) — compile-verified, matrix operand order checked by reading, not run on hardware
//
#include <hip/hip_runtime.h>

typedef unsigned short u16;
typedef __bf16 bf16_t;
typedef bf16_t v16bf __attribute__((ext_vector_type(16)));
typedef float  v8f   __attribute__((ext_vector_type(8)));
typedef unsigned uint4v __attribute__((ext_vector_type(4)));
typedef unsigned uint8v __attribute__((ext_vector_type(8)));

#define SCALEF 0.08838834764831845f   // 128^-0.5
#define NEGF   (-1.0e9f)

union FragU { v16bf v; uint4 q[2]; };

__device__ __forceinline__ u16 f2bf(float x) {
  unsigned u = __float_as_uint(x);
  u += 0x7fffu + ((u >> 16) & 1u);
  return (u16)(u >> 16);
}

// Load a 16x32 bf16 fragment (A layout; also valid as "B via row=n of W^T").
// tile rows are K-contiguous with leading dim `ld` (elements).
__device__ __forceinline__ v16bf load_frag(const u16* tile, int row, int ld, int k0, int lane) {
  int r  = row + (lane & 15);
  int kg = (lane >> 4) << 3;            // 0 or 8
  const u16* p = tile + r * ld + k0 + kg;
  FragU f;
  f.q[0] = *(const uint4*)(p);          // K = k0 + kg + 0..7
  f.q[1] = *(const uint4*)(p + 16);     // K = k0 + 16 + kg + 0..7
  return f.v;
}

#define WMMA_BF16(a, b, c) \
  __builtin_amdgcn_wmma_f32_16x16x32_bf16(false, (a), false, (b), (short)0, (c), false, false)

// ------------------------------------------------------- TDM (CDNA5 DMA)
// Issue a 2D tensor tile DMA: global (row-major, 2-byte elems, row stride
// `stride_elems`) -> LDS at byte offset lds_byte_off. tile = tile_d1 rows
// x tile_d0 elems. Wave-level op, tracked with TENSORcnt.
__device__ __forceinline__ void tdm_load_tile_2d(const void* gaddr, unsigned lds_byte_off,
                                                 unsigned tensor_d0, unsigned tensor_d1,
                                                 unsigned tile_d0, unsigned tile_d1,
                                                 unsigned stride_elems) {
  unsigned long long ga = (unsigned long long)gaddr;
  uint4v g0;
  g0[0] = 1u;                                                  // count=1, user descriptor
  g0[1] = lds_byte_off;                                        // lds_addr (bytes)
  g0[2] = (unsigned)ga;                                        // global_addr[31:0]
  g0[3] = (unsigned)((ga >> 32) & 0x01FFFFFFu) | (2u << 30);   // addr[56:32] | type=2
  uint8v g1;
  g1[0] = 0x00010000u;                                         // data_size=1 (2 bytes)
  g1[1] = (tensor_d0 & 0xFFFFu) << 16;                         // tensor_dim0[15:0]
  g1[2] = ((tensor_d0 >> 16) & 0xFFFFu) | ((tensor_d1 & 0xFFFFu) << 16);
  g1[3] = ((tensor_d1 >> 16) & 0xFFFFu) | ((tile_d0 & 0xFFFFu) << 16);
  g1[4] = (tile_d1 & 0xFFFFu);                                 // tile_dim1 | tile_dim2=0
  g1[5] = stride_elems;                                        // tensor_dim0_stride[31:0]
  g1[6] = 0u;
  g1[7] = 0u;
  asm volatile("tensor_load_to_lds %0, %1" :: "s"(g0), "s"(g1) : "memory");
}

// ---------------------------------------------------------------- converts
__global__ void cvt_bf16(const float* __restrict__ x, u16* __restrict__ y, int n) {
  for (int i = blockIdx.x * blockDim.x + threadIdx.x; i < n; i += gridDim.x * blockDim.x)
    y[i] = f2bf(x[i]);
}

// w[K][N] f32 -> wt[N][K] bf16 (tiled transpose)
__global__ __launch_bounds__(256) void wtrans(const float* __restrict__ w, u16* __restrict__ wt,
                                              int K, int N) {
  __shared__ float t[32][33];
  int kb = blockIdx.x * 32, nb = blockIdx.y * 32;
  int x = threadIdx.x, y = threadIdx.y;           // 32 x 8
  for (int i = y; i < 32; i += 8) t[i][x] = w[(size_t)(kb + i) * N + nb + x];
  __syncthreads();
  for (int i = y; i < 32; i += 8) wt[(size_t)(nb + i) * K + kb + x] = f2bf(t[x][i]);
}

// v[S][HKD] f32 -> vt[HKD][S] bf16
__global__ void vtrans(const float* __restrict__ v, u16* __restrict__ vt, int S, int HKD, int n) {
  for (int i = blockIdx.x * blockDim.x + threadIdx.x; i < n; i += gridDim.x * blockDim.x) {
    int s = i % S;
    int c = i / S;
    vt[i] = f2bf(v[(size_t)s * HKD + c]);
  }
}

// -------------------------------------------------------------------- GEMM
// C[M][N] f32 = A[M][K] bf16 @ Bt[N][K]^T bf16.  128x128 tile, K-step 64.
// Tiles are streamed into LDS by the Tensor Data Mover, double-buffered:
// wave 0 issues tensor_load_to_lds for tile i+1 while all waves compute
// tile i; s_wait_tensorcnt + workgroup barrier flips buffers.
// Dynamic LDS: 4 x 16KB = 64KB  (As0 | Bs0 | As1 | Bs1).
__global__ __launch_bounds__(256) void gemm_bf16(const u16* __restrict__ A,
                                                 const u16* __restrict__ Bt,
                                                 float* __restrict__ C,
                                                 int M, int N, int K) {
  extern __shared__ u16 smem[];
  int tid = threadIdx.x;
  int lane = tid & 31, wave = tid >> 5;
  int lh = lane >> 4, nl = lane & 15;
  int m0 = (wave >> 1) * 32;
  int n0 = (wave & 1) * 64;
  size_t bm = (size_t)blockIdx.y * 128, bn = (size_t)blockIdx.x * 128;
  v8f acc[2][4] = {};

  const int nk = K >> 6;
  if (wave == 0) {
    tdm_load_tile_2d(A + bm * K, 0u, (unsigned)K, (unsigned)(M - bm), 64u, 128u, (unsigned)K);
    tdm_load_tile_2d(Bt + bn * K, 16384u, (unsigned)K, (unsigned)(N - bn), 64u, 128u, (unsigned)K);
    __builtin_amdgcn_s_wait_tensorcnt(0);
  }
  __syncthreads();

  for (int it = 0; it < nk; ++it) {
    int buf = it & 1;
    const u16* As = smem + buf * 16384;       // element offsets: 0 / 16384
    const u16* Bs = As + 8192;
    if (wave == 0 && it + 1 < nk) {
      int k0n = (it + 1) << 6;
      unsigned dstb = (unsigned)(1 - buf) * 32768u;
      tdm_load_tile_2d(A + bm * K + k0n, dstb,
                       (unsigned)(K - k0n), (unsigned)(M - bm), 64u, 128u, (unsigned)K);
      tdm_load_tile_2d(Bt + bn * K + k0n, dstb + 16384u,
                       (unsigned)(K - k0n), (unsigned)(N - bn), 64u, 128u, (unsigned)K);
    }
    for (int kk = 0; kk < 64; kk += 32) {
      v16bf a0 = load_frag(As, m0, 64, kk, lane);
      v16bf a1 = load_frag(As, m0 + 16, 64, kk, lane);
      for (int j = 0; j < 4; ++j) {
        v16bf b = load_frag(Bs, n0 + j * 16, 64, kk, lane);
        acc[0][j] = WMMA_BF16(a0, b, acc[0][j]);
        acc[1][j] = WMMA_BF16(a1, b, acc[1][j]);
      }
    }
    if (wave == 0) __builtin_amdgcn_s_wait_tensorcnt(0);
    __syncthreads();
  }
  for (int i2 = 0; i2 < 2; ++i2)
    for (int j = 0; j < 4; ++j)
      for (int r = 0; r < 8; ++r) {
        size_t gm = bm + m0 + i2 * 16 + r + 8 * lh;
        size_t gn = bn + n0 + j * 16 + nl;
        C[gm * N + gn] = acc[i2][j][r];
      }
}

// ------------------------------------------------------------ RMS (+RoPE)
// x[S][H*128] f32 -> outb/outf [H][S][128]
__global__ __launch_bounds__(128) void rmsnorm(const float* __restrict__ x,
                                               const float* __restrict__ w,
                                               const int* __restrict__ positions,
                                               u16* __restrict__ outb, float* __restrict__ outf,
                                               int S, int H, int do_rope) {
  int s = blockIdx.x, h = blockIdx.y, d = threadIdx.x;
  __shared__ float red[4];
  __shared__ float buf[128];
  float v = x[(size_t)s * H * 128 + h * 128 + d];
  float ss = v * v;
  for (int off = 1; off < 32; off <<= 1) ss += __shfl_xor(ss, off, 32);
  if ((d & 31) == 0) red[d >> 5] = ss;
  __syncthreads();
  float tot = red[0] + red[1] + red[2] + red[3];
  float y = v * rsqrtf(tot * (1.0f / 128.0f) + 1e-6f) * w[d];
  if (do_rope) {
    buf[d] = y;
    __syncthreads();
    int i = d & 63;
    float pos = (float)positions[s];
    float ang = pos * __expf(-(float)i * (13.815510557964274f / 64.0f));  // theta=1e6
    float c = cosf(ang), sn = sinf(ang);
    float y1 = buf[i], y2 = buf[i + 64];
    y = (d < 64) ? (y1 * c - y2 * sn) : (y2 * c + y1 * sn);
  }
  size_t o = ((size_t)h * S + s) * 128 + d;
  if (outb) outb[o] = f2bf(y);
  if (outf) outf[o] = y;
}

// lmk[n][kh][128] = mean over 64 rows of hk_f[kh][n*64+j][128]
__global__ __launch_bounds__(128) void lmk_mean(const float* __restrict__ hkf,
                                                float* __restrict__ lmkf, int S) {
  int n = blockIdx.x, kh = blockIdx.y, d = threadIdx.x;
  float a = 0.f;
  for (int j = 0; j < 64; ++j) a += hkf[((size_t)kh * S + n * 64 + j) * 128 + d];
  lmkf[(n * 2 + kh) * 128 + d] = a * (1.0f / 64.0f);
}

// ------------------------------------------------------------- SWA (flash)
__global__ __launch_bounds__(128) void swa_attn(const u16* __restrict__ qg,   // [24][S][128]
                                                const u16* __restrict__ kg,   // [6][S][128]
                                                const u16* __restrict__ vtg,  // [6][128][S]
                                                u16* __restrict__ attn,       // [S][4096]
                                                int S) {
  __shared__ u16 q_s[64 * 128];
  __shared__ u16 k_s[64 * 128];
  __shared__ u16 vT_s[128 * 64];
  __shared__ u16 p_s[4][16 * 64];
  int tid = threadIdx.x;
  int lane = tid & 31, wave = tid >> 5;
  int lh = lane >> 4, nl = lane & 15;
  int qs0 = blockIdx.x * 64;
  int h = blockIdx.y;
  int kh = h >> 2;
  {
    const uint4* src = (const uint4*)(qg + ((size_t)h * S + qs0) * 128);
    uint4* dst = (uint4*)q_s;
    for (int i = tid; i < 1024; i += 128) dst[i] = src[i];
  }
  int m0 = wave * 16;
  v8f O[8] = {};
  float mrow[8], lrow[8];
  for (int r = 0; r < 8; ++r) { mrow[r] = -3.0e38f; lrow[r] = 0.f; }
  int lo = qs0 - 511; if (lo < 0) lo = 0;
  for (int kt = lo / 64; kt <= (int)blockIdx.x; ++kt) {
    __syncthreads();
    {
      const uint4* src = (const uint4*)(kg + ((size_t)kh * S + kt * 64) * 128);
      uint4* dst = (uint4*)k_s;
      for (int i = tid; i < 1024; i += 128) dst[i] = src[i];
    }
    {
      const u16* vb = vtg + (size_t)kh * 128 * S;
      for (int i = tid; i < 1024; i += 128) {
        int row = i >> 3, c = (i & 7) * 8;
        *(uint4*)&vT_s[row * 64 + c] = *(const uint4*)&vb[(size_t)row * S + kt * 64 + c];
      }
    }
    __syncthreads();
    v8f sc[4] = {};
    for (int ks = 0; ks < 4; ++ks) {
      v16bf a = load_frag(q_s, m0, 128, ks * 32, lane);
      for (int n = 0; n < 4; ++n) {
        v16bf b = load_frag(k_s, n * 16, 128, ks * 32, lane);
        sc[n] = WMMA_BF16(a, b, sc[n]);
      }
    }
    for (int r = 0; r < 8; ++r) {
      int qrow = qs0 + m0 + r + 8 * lh;
      float mx = -3.0e38f;
      for (int n = 0; n < 4; ++n) {
        int key = kt * 64 + n * 16 + nl;
        float v2 = sc[n][r] * SCALEF;
        bool ok = (key <= qrow) && (qrow - key < 512);
        v2 = ok ? v2 : NEGF;
        sc[n][r] = v2;
        mx = fmaxf(mx, v2);
      }
      for (int off = 1; off < 16; off <<= 1) mx = fmaxf(mx, __shfl_xor(mx, off, 32));
      float mn = fmaxf(mrow[r], mx);
      float al = __expf(mrow[r] - mn);
      lrow[r] *= al;
      for (int j = 0; j < 8; ++j) O[j][r] *= al;
      mrow[r] = mn;
      float ps = 0.f;
      for (int n = 0; n < 4; ++n) {
        float p = __expf(sc[n][r] - mn);
        ps += p;
        p_s[wave][(r + 8 * lh) * 64 + n * 16 + nl] = f2bf(p);
      }
      for (int off = 1; off < 16; off <<= 1) ps += __shfl_xor(ps, off, 32);
      lrow[r] += ps;
    }
    for (int ks = 0; ks < 2; ++ks) {
      v16bf a = load_frag(p_s[wave], 0, 64, ks * 32, lane);
      for (int j = 0; j < 8; ++j) {
        v16bf b = load_frag(vT_s, j * 16, 64, ks * 32, lane);
        O[j] = WMMA_BF16(a, b, O[j]);
      }
    }
  }
  for (int r = 0; r < 8; ++r) {
    int s = qs0 + m0 + r + 8 * lh;
    float inv = 1.0f / lrow[r];
    for (int j = 0; j < 8; ++j)
      attn[(size_t)s * 4096 + h * 128 + j * 16 + nl] = f2bf(O[j][r] * inv);
  }
}

// --------------------------------------------------------------- HSA
__global__ __launch_bounds__(128) void hsa_attn(const u16* __restrict__ hqg,   // [8][S][128]
                                                const u16* __restrict__ hkg,   // [2][S][128]
                                                const u16* __restrict__ hvtg,  // [2][128][S]
                                                const float* __restrict__ selq, // [2][S][128]
                                                const float* __restrict__ lmkf, // [16][2][128]
                                                const float* __restrict__ gproj,// [S][1024]
                                                u16* __restrict__ attn,
                                                const int* __restrict__ positions, int S) {
  __shared__ u16 q_s[64 * 128];
  __shared__ u16 k_s[64 * 128];
  __shared__ u16 vT_s[128 * 64];
  __shared__ u16 p_s[4][16 * 64];
  __shared__ float pw_s[64][16];
  __shared__ int flag[16];
  int tid = threadIdx.x;
  int lane = tid & 31, wave = tid >> 5;
  int lh = lane >> 4, nl = lane & 15;
  int qs0 = blockIdx.x * 64;
  int h = blockIdx.y;
  int kh = h >> 2;

  for (int i = tid; i < 64 * 16; i += 128) ((float*)pw_s)[i] = 0.f;
  if (tid < 16) flag[tid] = 0;
  __syncthreads();
  {
    const uint4* src = (const uint4*)(hqg + ((size_t)h * S + qs0) * 128);
    uint4* dst = (uint4*)q_s;
    for (int i = tid; i < 1024; i += 128) dst[i] = src[i];
  }
  if (tid < 64) {
    int s = qs0 + tid;
    int pos = positions[s];
    const float* qv = selq + ((size_t)kh * S + s) * 128;
    float ls[16];
    for (int n = 0; n < 16; ++n) {
      const float* lv = lmkf + (n * 2 + kh) * 128;
      float dot = 0.f;
      for (int d = 0; d < 128; ++d) dot += qv[d] * lv[d];
      ls[n] = (n * 64 <= pos) ? dot * SCALEF : NEGF;
    }
    float vals[8]; int ids[8]; bool used[16];
    for (int n = 0; n < 16; ++n) used[n] = false;
    for (int k2 = 0; k2 < 8; ++k2) {
      float best = -3.4e38f; int bi = 0;
      for (int n = 0; n < 16; ++n)
        if (!used[n] && ls[n] > best) { best = ls[n]; bi = n; }
      used[bi] = true; vals[k2] = best; ids[k2] = bi;
    }
    float mx = vals[0], sum = 0.f, e[8];
    for (int k2 = 0; k2 < 8; ++k2) { e[k2] = __expf(vals[k2] - mx); sum += e[k2]; }
    float inv = 1.0f / sum;
    for (int k2 = 0; k2 < 8; ++k2) pw_s[tid][ids[k2]] = e[k2] * inv;
  }
  __syncthreads();
  if (tid < 16) {
    int f = 0;
    for (int t = 0; t < 64; ++t) f |= (pw_s[t][tid] > 0.f) ? 1 : 0;
    flag[tid] = f;
  }
  __syncthreads();

  int m0 = wave * 16;
  v8f O[8] = {};
  for (int nb = 0; nb <= (int)blockIdx.x; ++nb) {
    if (!flag[nb]) continue;   // uniform across block
    __syncthreads();
    {
      const uint4* src = (const uint4*)(hkg + ((size_t)kh * S + nb * 64) * 128);
      uint4* dst = (uint4*)k_s;
      for (int i = tid; i < 1024; i += 128) dst[i] = src[i];
    }
    {
      const u16* vb = hvtg + (size_t)kh * 128 * S;
      for (int i = tid; i < 1024; i += 128) {
        int row = i >> 3, c = (i & 7) * 8;
        *(uint4*)&vT_s[row * 64 + c] = *(const uint4*)&vb[(size_t)row * S + nb * 64 + c];
      }
    }
    __syncthreads();
    v8f sc[4] = {};
    for (int ks = 0; ks < 4; ++ks) {
      v16bf a = load_frag(q_s, m0, 128, ks * 32, lane);
      for (int n = 0; n < 4; ++n) {
        v16bf b = load_frag(k_s, n * 16, 128, ks * 32, lane);
        sc[n] = WMMA_BF16(a, b, sc[n]);
      }
    }
    for (int r = 0; r < 8; ++r) {
      int qrow = qs0 + m0 + r + 8 * lh;
      float mx = -3.0e38f;
      for (int n = 0; n < 4; ++n) {
        int key = nb * 64 + n * 16 + nl;
        float v2 = sc[n][r] * SCALEF;
        v2 = (key <= qrow) ? v2 : NEGF;
        sc[n][r] = v2;
        mx = fmaxf(mx, v2);
      }
      for (int off = 1; off < 16; off <<= 1) mx = fmaxf(mx, __shfl_xor(mx, off, 32));
      float ssum = 0.f;
      for (int n = 0; n < 4; ++n) {
        float p = __expf(sc[n][r] - mx);
        ssum += p;
        sc[n][r] = p;
      }
      for (int off = 1; off < 16; off <<= 1) ssum += __shfl_xor(ssum, off, 32);
      float pf = pw_s[m0 + r + 8 * lh][nb] / ssum;
      for (int n = 0; n < 4; ++n)
        p_s[wave][(r + 8 * lh) * 64 + n * 16 + nl] = f2bf(sc[n][r] * pf);
    }
    for (int ks = 0; ks < 2; ++ks) {
      v16bf a = load_frag(p_s[wave], 0, 64, ks * 32, lane);
      for (int j = 0; j < 8; ++j) {
        v16bf b = load_frag(vT_s, j * 16, 64, ks * 32, lane);
        O[j] = WMMA_BF16(a, b, O[j]);
      }
    }
  }
  for (int r = 0; r < 8; ++r) {
    int s = qs0 + m0 + r + 8 * lh;
    for (int j = 0; j < 8; ++j) {
      int col = j * 16 + nl;
      float g = gproj[(size_t)s * 1024 + h * 128 + col];
      float sig = 1.0f / (1.0f + __expf(-g));
      attn[(size_t)s * 4096 + 3072 + h * 128 + col] = f2bf(O[j][r] * sig);
    }
  }
}

// ------------------------------------------------------------------ launch
extern "C" void kernel_launch(void* const* d_in, const int* in_sizes, int n_in,
                              void* d_out, int out_size, void* d_ws, size_t ws_size,
                              hipStream_t stream) {
  (void)in_sizes; (void)n_in; (void)out_size; (void)ws_size;
  const int S = 1024, H = 4096;
  const int* positions  = (const int*)d_in[0];
  const float* hs       = (const float*)d_in[1];
  const float* wq       = (const float*)d_in[2];
  const float* wk       = (const float*)d_in[3];
  const float* wv       = (const float*)d_in[4];
  const float* hsa_wq   = (const float*)d_in[5];
  const float* hsa_wk   = (const float*)d_in[6];
  const float* hsa_wv   = (const float*)d_in[7];
  const float* lmk_wq   = (const float*)d_in[8];
  const float* gate_w   = (const float*)d_in[9];
  const float* wo       = (const float*)d_in[10];
  const float* q_norm_w = (const float*)d_in[11];
  const float* k_norm_w = (const float*)d_in[12];
  const float* lmk_norm_w = (const float*)d_in[13];

  char* ws = (char*)d_ws;
  size_t off = 0;
  auto alloc = [&](size_t bytes) -> void* {
    void* p = ws + off;
    off += (bytes + 255) & ~(size_t)255;
    return p;
  };
  u16* hs_bf  = (u16*)alloc((size_t)S * H * 2);
  u16* wqT    = (u16*)alloc((size_t)H * 3072 * 2);
  u16* wkT    = (u16*)alloc((size_t)H * 768 * 2);
  u16* wvT    = (u16*)alloc((size_t)H * 768 * 2);
  u16* hwqT   = (u16*)alloc((size_t)H * 1024 * 2);
  u16* hwkT   = (u16*)alloc((size_t)H * 256 * 2);
  u16* hwvT   = (u16*)alloc((size_t)H * 256 * 2);
  u16* lmkT   = (u16*)alloc((size_t)H * 256 * 2);
  u16* gateT  = (u16*)alloc((size_t)H * 1024 * 2);
  u16* woT    = (u16*)alloc((size_t)4096 * H * 2);
  float* qproj   = (float*)alloc((size_t)S * 3072 * 4);
  float* kproj   = (float*)alloc((size_t)S * 768 * 4);
  float* vproj   = (float*)alloc((size_t)S * 768 * 4);
  float* hqproj  = (float*)alloc((size_t)S * 1024 * 4);
  float* hkproj  = (float*)alloc((size_t)S * 256 * 4);
  float* hvproj  = (float*)alloc((size_t)S * 256 * 4);
  float* lmkproj = (float*)alloc((size_t)S * 256 * 4);
  float* gproj   = (float*)alloc((size_t)S * 1024 * 4);
  u16* q_bf   = (u16*)alloc((size_t)24 * S * 128 * 2);
  u16* k_bf   = (u16*)alloc((size_t)6 * S * 128 * 2);
  u16* vT_bf  = (u16*)alloc((size_t)6 * 128 * S * 2);
  u16* hq_bf  = (u16*)alloc((size_t)8 * S * 128 * 2);
  u16* hk_bf  = (u16*)alloc((size_t)2 * S * 128 * 2);
  float* hk_f = (float*)alloc((size_t)2 * S * 128 * 4);
  u16* hvT_bf = (u16*)alloc((size_t)2 * 128 * S * 2);
  float* selq_f = (float*)alloc((size_t)2 * S * 128 * 4);
  float* lmk_f  = (float*)alloc((size_t)16 * 2 * 128 * 4);
  u16* attn_bf = (u16*)alloc((size_t)S * 4096 * 2);

  cvt_bf16<<<2048, 256, 0, stream>>>(hs, hs_bf, S * H);

  dim3 tb(32, 8);
  wtrans<<<dim3(H / 32, 3072 / 32), tb, 0, stream>>>(wq, wqT, H, 3072);
  wtrans<<<dim3(H / 32, 768 / 32),  tb, 0, stream>>>(wk, wkT, H, 768);
  wtrans<<<dim3(H / 32, 768 / 32),  tb, 0, stream>>>(wv, wvT, H, 768);
  wtrans<<<dim3(H / 32, 1024 / 32), tb, 0, stream>>>(hsa_wq, hwqT, H, 1024);
  wtrans<<<dim3(H / 32, 256 / 32),  tb, 0, stream>>>(hsa_wk, hwkT, H, 256);
  wtrans<<<dim3(H / 32, 256 / 32),  tb, 0, stream>>>(hsa_wv, hwvT, H, 256);
  wtrans<<<dim3(H / 32, 256 / 32),  tb, 0, stream>>>(lmk_wq, lmkT, H, 256);
  wtrans<<<dim3(H / 32, 1024 / 32), tb, 0, stream>>>(gate_w, gateT, H, 1024);
  wtrans<<<dim3(4096 / 32, H / 32), tb, 0, stream>>>(wo, woT, 4096, H);

  const size_t gemm_lds = 65536;  // 2 x (A 16KB + B 16KB) double buffer
  gemm_bf16<<<dim3(3072 / 128, S / 128), 256, gemm_lds, stream>>>(hs_bf, wqT, qproj, S, 3072, H);
  gemm_bf16<<<dim3(768 / 128,  S / 128), 256, gemm_lds, stream>>>(hs_bf, wkT, kproj, S, 768, H);
  gemm_bf16<<<dim3(768 / 128,  S / 128), 256, gemm_lds, stream>>>(hs_bf, wvT, vproj, S, 768, H);
  gemm_bf16<<<dim3(1024 / 128, S / 128), 256, gemm_lds, stream>>>(hs_bf, hwqT, hqproj, S, 1024, H);
  gemm_bf16<<<dim3(256 / 128,  S / 128), 256, gemm_lds, stream>>>(hs_bf, hwkT, hkproj, S, 256, H);
  gemm_bf16<<<dim3(256 / 128,  S / 128), 256, gemm_lds, stream>>>(hs_bf, hwvT, hvproj, S, 256, H);
  gemm_bf16<<<dim3(256 / 128,  S / 128), 256, gemm_lds, stream>>>(hs_bf, lmkT, lmkproj, S, 256, H);
  gemm_bf16<<<dim3(1024 / 128, S / 128), 256, gemm_lds, stream>>>(hs_bf, gateT, gproj, S, 1024, H);

  rmsnorm<<<dim3(S, 24), 128, 0, stream>>>(qproj, q_norm_w, positions, q_bf, nullptr, S, 24, 1);
  rmsnorm<<<dim3(S, 6),  128, 0, stream>>>(kproj, k_norm_w, positions, k_bf, nullptr, S, 6, 1);
  rmsnorm<<<dim3(S, 8),  128, 0, stream>>>(hqproj, q_norm_w, positions, hq_bf, nullptr, S, 8, 0);
  rmsnorm<<<dim3(S, 2),  128, 0, stream>>>(hkproj, k_norm_w, positions, hk_bf, hk_f, S, 2, 0);
  rmsnorm<<<dim3(S, 2),  128, 0, stream>>>(lmkproj, lmk_norm_w, positions, nullptr, selq_f, S, 2, 0);

  vtrans<<<1024, 256, 0, stream>>>(vproj, vT_bf, S, 6 * 128, 6 * 128 * S);
  vtrans<<<512,  256, 0, stream>>>(hvproj, hvT_bf, S, 2 * 128, 2 * 128 * S);
  lmk_mean<<<dim3(16, 2), 128, 0, stream>>>(hk_f, lmk_f, S);

  swa_attn<<<dim3(S / 64, 24), 128, 0, stream>>>(q_bf, k_bf, vT_bf, attn_bf, S);
  hsa_attn<<<dim3(S / 64, 8), 128, 0, stream>>>(hq_bf, hk_bf, hvT_bf, selq_f, lmk_f, gproj,
                                                attn_bf, positions, S);

  gemm_bf16<<<dim3(4096 / 128, S / 128), 256, gemm_lds, stream>>>(attn_bf, woT, (float*)d_out,
                                                                  S, 4096, H);
}
